// AdditiveResonanceAttentionV17_47648367182291
// MI455X (gfx1250) — matrix operand order repaired and checked by reading, not compiled, not run on hardware
//
#include <hip/hip_runtime.h>
#include <hip/hip_bf16.h>
#include <math.h>
#include <stdint.h>

#define BQ 4
#define LQ 2048
#define DQ 512
#define HQ 8
#define DHQ 64
#define DHP 96   // padded head dim: 64 semantic + cos/sin/mask + zero pad to 3x32

// 1 = stage K'/V tiles with global_load_async_to_lds_b128 (ASYNCcnt path)
// 0 = sync fallback (global load -> ds store)
#define USE_ASYNC_COPY 1

typedef __attribute__((ext_vector_type(16))) _Float16 v16h;
typedef __attribute__((ext_vector_type(8)))  _Float16 v8h;
typedef __attribute__((ext_vector_type(8)))  float    v8f;

static __device__ __forceinline__ v16h frag_ld(const _Float16* p0, const _Float16* p1) {
  v8h lo = *(const v8h*)p0;
  v8h hi = *(const v8h*)p1;
  v16h r;
#pragma unroll
  for (int e = 0; e < 8; ++e) { r[e] = lo[e]; r[e + 8] = hi[e]; }
  return r;
}

static __device__ __forceinline__ v8f wmma_f16(v16h a, v16h b, v8f c) {
  return __builtin_amdgcn_wmma_f32_16x16x32_f16(false, a, false, b, (short)0, c, false, false);
}

// ---- CDNA5 async global->LDS copy (16B per lane), tracked by ASYNCcnt ----
static __device__ __forceinline__ void cp16(_Float16* ldst, const _Float16* gsrc) {
#if USE_ASYNC_COPY
  // VDST = 32-bit LDS offset (low 32 bits of generic LDS pointer),
  // VADDR = 64-bit global address (GV mode).
  asm volatile("global_load_async_to_lds_b128 %0, %1, off"
               :: "v"((uint32_t)(uintptr_t)ldst), "v"(gsrc)
               : "memory");
#else
  *(v8h*)ldst = *(const v8h*)gsrc;
#endif
}

static __device__ __forceinline__ void wait_async0() {
#if USE_ASYNC_COPY
#if __has_builtin(__builtin_amdgcn_s_wait_asynccnt)
  __builtin_amdgcn_s_wait_asynccnt(0);
#else
  asm volatile("s_wait_asynccnt 0x0" ::: "memory");
#endif
#endif
}

// ---------------------------------------------------------------- converts
__global__ void cvt_f32_to_f16(const float* __restrict__ src,
                               _Float16* __restrict__ dst, int n) {
  int i = blockIdx.x * blockDim.x + threadIdx.x;
  if (i < n) dst[i] = (_Float16)src[i];
}

// ------------------------------------------------- resonance / mask fold-in
// gamma*mag_i*mag_j*cos(phi_i-phi_j) = rank-2 outer product; folded into the
// score GEMM via padded dims 64/65; additive mask folded into dim 66.
__global__ void resonance_fill(const float* __restrict__ phi,
                               const float* __restrict__ mag,
                               const float* __restrict__ mask,
                               const float* __restrict__ gamma,
                               _Float16* __restrict__ Qh,
                               _Float16* __restrict__ Kh) {
  int t = blockIdx.x * 256 + threadIdx.x;       // over B*H*L
  if (t >= BQ * HQ * LQ) return;
  int j  = t & (LQ - 1);
  int bh = t >> 11;
  int h  = bh & (HQ - 1);
  int b  = bh >> 3;
  float g  = gamma[h];
  float sg = (g < 0.f) ? -1.f : 1.f;
  float sc = sqrtf(fabsf(g));
  float c = __cosf(phi[t]);
  float s = __sinf(phi[t]);
  float m = mag[t];
  _Float16* q = Qh + (size_t)t * DHP;
  _Float16* k = Kh + (size_t)t * DHP;
  q[64] = (_Float16)(sg * sc * m * c);
  q[65] = (_Float16)(sg * sc * m * s);
  q[66] = (_Float16)1.0f;
  k[64] = (_Float16)(sc * m * c);
  k[65] = (_Float16)(sc * m * s);
  k[66] = (_Float16)(mask[b * LQ + j]);
#pragma unroll
  for (int d2 = 67; d2 < DHP; ++d2) { q[d2] = (_Float16)0.f; k[d2] = (_Float16)0.f; }
}

// ------------------------------------------------------------- QKV GEMM
// 128x64 output tile per block (8 waves, each 16 rows x 64 cols = 4 accs).
// Q rows pre-scaled by 1/sqrt(DH)=1/8; V stored transposed (B,H,DH,L).
__global__ __launch_bounds__(256) void qkv_gemm(
    const _Float16* __restrict__ X,
    const _Float16* __restrict__ Wq, const _Float16* __restrict__ Wk,
    const _Float16* __restrict__ Wv,
    const float* __restrict__ bq, const float* __restrict__ bk,
    const float* __restrict__ bv,
    _Float16* __restrict__ Qh, _Float16* __restrict__ Kh,
    _Float16* __restrict__ Vt) {
  const int lane = threadIdx.x & 31;
  const int w    = threadIdx.x >> 5;
  const int z    = blockIdx.z;
  const int m0   = blockIdx.x * 128 + w * 16;
  const int n0   = blockIdx.y * 64;
  const _Float16* W    = (z == 0) ? Wq : (z == 1) ? Wk : Wv;
  const float*    bias = (z == 0) ? bq : (z == 1) ? bk : bv;
  const int mrow = lane & 15, sel = lane >> 4;

  v8f acc[4];
#pragma unroll
  for (int nt = 0; nt < 4; ++nt)
#pragma unroll
    for (int r = 0; r < 8; ++r) acc[nt][r] = 0.f;

  for (int kc = 0; kc < DQ / 32; ++kc) {
    const _Float16* ap = X + (size_t)(m0 + mrow) * DQ + kc * 32 + 8 * sel;
    if (kc + 2 < DQ / 32) {
      __builtin_prefetch(ap + 64, 0, 1);                       // -> global_prefetch_b8
      __builtin_prefetch(W + (size_t)(n0 + mrow) * DQ + kc * 32 + 64, 0, 1);
    }
    v16h af = frag_ld(ap, ap + 16);
#pragma unroll
    for (int nt = 0; nt < 4; ++nt) {
      const _Float16* bp = W + (size_t)(n0 + nt * 16 + mrow) * DQ + kc * 32 + 8 * sel;
      v16h bf = frag_ld(bp, bp + 16);
      acc[nt] = wmma_f16(af, bf, acc[nt]);
    }
  }
#pragma unroll
  for (int nt = 0; nt < 4; ++nt)
#pragma unroll
    for (int r = 0; r < 8; ++r) {
      int gm = m0 + r + 8 * sel;        // row in [0, B*L)
      int gn = n0 + nt * 16 + mrow;     // col in [0, D)
      float v = acc[nt][r] + bias[gn];
      int b = gm >> 11, li = gm & (LQ - 1);
      int h = gn >> 6,  dh = gn & 63;
      int bh = b * HQ + h;
      if (z == 0) {
        Qh[((size_t)bh * LQ + li) * DHP + dh] = (_Float16)(v * 0.125f);
      } else if (z == 1) {
        Kh[((size_t)bh * LQ + li) * DHP + dh] = (_Float16)v;
      } else {
        Vt[((size_t)bh * DHQ + dh) * LQ + li] = (_Float16)v;   // transposed
      }
    }
}

// ---- cooperative async staging of one 64-column K'/V tile into LDS ----
static __device__ __forceinline__ void stage_tile(
    const _Float16* __restrict__ Kh, const _Float16* __restrict__ Vt,
    int bh, int j0, _Float16* kbuf, _Float16* vbuf, int tid) {
  // K' tile: rows j0..j0+63 x DHP halves -> one contiguous 12KB block
  const _Float16* kg = Kh + ((size_t)bh * LQ + j0) * DHP;
#pragma unroll
  for (int i = 0; i < 3; ++i) {
    int c = tid + 256 * i;              // 768 x 16B chunks
    cp16(kbuf + c * 8, kg + c * 8);
  }
  // V tile: 64 dh-rows x 64 j halves (transposed layout -> contiguous rows)
#pragma unroll
  for (int i = 0; i < 2; ++i) {
    int c  = tid + 256 * i;             // 512 x 16B chunks
    int dh = c >> 3, part = c & 7;
    cp16(vbuf + c * 8, Vt + ((size_t)bh * DHQ + dh) * LQ + j0 + part * 8);
  }
}

// -------------------------------------------------------- flash attention
// Block = (row-tile of 128, bh). K'/V column tiles are double-buffered in
// LDS via async global->LDS DMA; wave owns 16 rows with online softmax.
__global__ __launch_bounds__(256) void flash_attn(
    const _Float16* __restrict__ Qh, const _Float16* __restrict__ Kh,
    const _Float16* __restrict__ Vt, _Float16* __restrict__ ctx16) {
  __shared__ _Float16 skt[2][64 * DHP];   // 24 KB
  __shared__ _Float16 svt[2][64 * DHQ];   // 16 KB
  __shared__ _Float16 pbuf[8 * 16 * 64];  // 16 KB (per-wave P staging)
  const int tid  = threadIdx.x;
  const int lane = tid & 31;
  const int w    = tid >> 5;
  const int it   = blockIdx.x;            // 0..15
  const int bh   = blockIdx.y;            // 0..31
  const int b    = bh >> 3, h = bh & 7;
  const int i0   = it * 128 + w * 16;
  const int mrow = lane & 15, sel = lane >> 4;

  v16h qa[3];
#pragma unroll
  for (int kc = 0; kc < 3; ++kc) {
    const _Float16* p = Qh + ((size_t)bh * LQ + i0 + mrow) * DHP + kc * 32 + 8 * sel;
    qa[kc] = frag_ld(p, p + 16);
  }

  float mrun[8], lrun[8];
  v8f cacc[4];
#pragma unroll
  for (int r = 0; r < 8; ++r) { mrun[r] = -3.0e38f; lrun[r] = 0.f; }
#pragma unroll
  for (int dt = 0; dt < 4; ++dt)
#pragma unroll
    for (int r = 0; r < 8; ++r) cacc[dt][r] = 0.f;

  _Float16* pw = pbuf + w * (16 * 64);

  stage_tile(Kh, Vt, bh, 0, skt[0], svt[0], tid);   // prologue: tile 0

  for (int jt = 0; jt < LQ / 64; ++jt) {
    const int s = jt & 1;
    wait_async0();        // my chunks of tile jt have landed
    __syncthreads();      // everyone's chunks landed; slot s^1 free to overwrite
    if (jt + 1 < LQ / 64)
      stage_tile(Kh, Vt, bh, (jt + 1) * 64, skt[s ^ 1], svt[s ^ 1], tid);

    const _Float16* kb = skt[s];
    const _Float16* vb = svt[s];

    // ---- scores 16x64 = Q'(16x96) x K'^T(96x64); bias+mask already folded
    v8f sacc[4];
#pragma unroll
    for (int nt = 0; nt < 4; ++nt) {
#pragma unroll
      for (int r = 0; r < 8; ++r) sacc[nt][r] = 0.f;
#pragma unroll
      for (int kc = 0; kc < 3; ++kc) {
        const _Float16* bp = kb + (nt * 16 + mrow) * DHP + kc * 32 + 8 * sel;
        v16h bf = frag_ld(bp, bp + 16);
        sacc[nt] = wmma_f16(qa[kc], bf, sacc[nt]);
      }
    }
    // ---- online softmax (row r lives in reg r, 16 lanes per half-wave)
#pragma unroll
    for (int r = 0; r < 8; ++r) {
      float tm = sacc[0][r];
#pragma unroll
      for (int nt = 1; nt < 4; ++nt) tm = fmaxf(tm, sacc[nt][r]);
#pragma unroll
      for (int off = 1; off < 16; off <<= 1) tm = fmaxf(tm, __shfl_xor(tm, off, 32));
      float mn    = fmaxf(mrun[r], tm);
      float alpha = __expf(mrun[r] - mn);
      mrun[r] = mn;
      float rs = 0.f;
#pragma unroll
      for (int nt = 0; nt < 4; ++nt) {
        float p = __expf(sacc[nt][r] - mn);
        sacc[nt][r] = p;
        rs += p;
      }
#pragma unroll
      for (int off = 1; off < 16; off <<= 1) rs += __shfl_xor(rs, off, 32);
      lrun[r] = lrun[r] * alpha + rs;
#pragma unroll
      for (int dt = 0; dt < 4; ++dt) cacc[dt][r] *= alpha;
    }
    // ---- P (C-layout regs) -> per-wave LDS -> A-layout fragments
#pragma unroll
    for (int nt = 0; nt < 4; ++nt)
#pragma unroll
      for (int r = 0; r < 8; ++r)
        pw[(r + 8 * sel) * 64 + nt * 16 + mrow] = (_Float16)sacc[nt][r];
    // ---- ctx += P(16x64) x V(64x64)
#pragma unroll
    for (int kc2 = 0; kc2 < 2; ++kc2) {
      const _Float16* pp = pw + mrow * 64 + kc2 * 32 + 8 * sel;
      v16h pf = frag_ld(pp, pp + 16);
#pragma unroll
      for (int dt = 0; dt < 4; ++dt) {
        const _Float16* vp = vb + (dt * 16 + mrow) * DHQ + kc2 * 32 + 8 * sel;
        v16h vf = frag_ld(vp, vp + 16);
        cacc[dt] = wmma_f16(pf, vf, cacc[dt]);
      }
    }
  }
  // ---- normalize, store ctx f16 reassembled as (B*L, D)
#pragma unroll
  for (int dt = 0; dt < 4; ++dt)
#pragma unroll
    for (int r = 0; r < 8; ++r) {
      int li  = i0 + r + 8 * sel;
      int col = h * DHQ + dt * 16 + mrow;
      float v = cacc[dt][r] / lrun[r];
      ctx16[((size_t)b * LQ + li) * DQ + col] = (_Float16)v;
    }
}

// ------------------------------------------- out-proj + residual + LayerNorm
__global__ __launch_bounds__(256) void out_proj_ln(
    const _Float16* __restrict__ ctx16, const _Float16* __restrict__ Wo,
    const float* __restrict__ bo, const float* __restrict__ resid,
    const float* __restrict__ lnw, const float* __restrict__ lnb,
    float* __restrict__ out) {
  __shared__ float sdata[16 * 512];
  const int lane = threadIdx.x & 31;
  const int w    = threadIdx.x >> 5;
  const int m0   = blockIdx.x * 16;
  const int n0   = w * 64;
  const int mrow = lane & 15, sel = lane >> 4;

  v8f acc[4];
#pragma unroll
  for (int nt = 0; nt < 4; ++nt)
#pragma unroll
    for (int r = 0; r < 8; ++r) acc[nt][r] = 0.f;

  for (int kc = 0; kc < DQ / 32; ++kc) {
    const _Float16* ap = ctx16 + (size_t)(m0 + mrow) * DQ + kc * 32 + 8 * sel;
    if (kc + 2 < DQ / 32) __builtin_prefetch(ap + 64, 0, 1);
    v16h af = frag_ld(ap, ap + 16);
#pragma unroll
    for (int nt = 0; nt < 4; ++nt) {
      const _Float16* bp = Wo + (size_t)(n0 + nt * 16 + mrow) * DQ + kc * 32 + 8 * sel;
      v16h bf = frag_ld(bp, bp + 16);
      acc[nt] = wmma_f16(af, bf, acc[nt]);
    }
  }
#pragma unroll
  for (int nt = 0; nt < 4; ++nt)
#pragma unroll
    for (int r = 0; r < 8; ++r) {
      int row = r + 8 * sel;
      int n   = n0 + nt * 16 + mrow;
      int g   = m0 + row;
      sdata[row * DQ + n] = acc[nt][r] + bo[n] + resid[(size_t)g * DQ + n];
    }
  __syncthreads();
  // LayerNorm: wave w normalizes rows 2w and 2w+1 (full 32-lane reduction)
#pragma unroll
  for (int rr = 0; rr < 2; ++rr) {
    int row = w * 2 + rr;
    int g   = m0 + row;
    float s1 = 0.f, s2 = 0.f;
#pragma unroll
    for (int t = 0; t < 16; ++t) {
      float x = sdata[row * DQ + lane + t * 32];
      s1 += x; s2 += x * x;
    }
#pragma unroll
    for (int off = 1; off < 32; off <<= 1) {
      s1 += __shfl_xor(s1, off, 32);
      s2 += __shfl_xor(s2, off, 32);
    }
    float mu  = s1 * (1.f / 512.f);
    float var = s2 * (1.f / 512.f) - mu * mu;
    float inv = rsqrtf(var + 1e-12f);
#pragma unroll
    for (int t = 0; t < 16; ++t) {
      int n = lane + t * 32;
      float x = sdata[row * DQ + n];
      out[(size_t)g * DQ + n] = lnw[n] * (x - mu) * inv + lnb[n];
    }
  }
}

// ----------------------------------------------------------------- launch
extern "C" void kernel_launch(void* const* d_in, const int* in_sizes, int n_in,
                              void* d_out, int out_size, void* d_ws, size_t ws_size,
                              hipStream_t stream) {
  (void)in_sizes; (void)n_in; (void)out_size; (void)ws_size;
  const float* hs    = (const float*)d_in[0];
  const float* mask  = (const float*)d_in[1];
  const float* phi   = (const float*)d_in[2];
  const float* mag   = (const float*)d_in[3];
  const float* Wq    = (const float*)d_in[4];
  const float* bq    = (const float*)d_in[5];
  const float* Wk    = (const float*)d_in[6];
  const float* bk    = (const float*)d_in[7];
  const float* Wv    = (const float*)d_in[8];
  const float* bv    = (const float*)d_in[9];
  const float* Wo    = (const float*)d_in[10];
  const float* bo    = (const float*)d_in[11];
  const float* gamma = (const float*)d_in[12];
  const float* lnw   = (const float*)d_in[13];
  const float* lnb   = (const float*)d_in[14];
  float* out = (float*)d_out;

  // workspace carve-up (halves); total ~50 MB
  _Float16* ws = (_Float16*)d_ws;
  const size_t N_HID = (size_t)BQ * LQ * DQ;          // 4194304
  const size_t N_W   = (size_t)DQ * DQ;               // 262144
  _Float16* X16  = ws;
  _Float16* Wq16 = X16 + N_HID;
  _Float16* Wk16 = Wq16 + N_W;
  _Float16* Wv16 = Wk16 + N_W;
  _Float16* Wo16 = Wv16 + N_W;
  _Float16* Qh   = Wo16 + N_W;
  _Float16* Kh   = Qh + (size_t)BQ * HQ * LQ * DHP;
  _Float16* Vt   = Kh + (size_t)BQ * HQ * LQ * DHP;
  _Float16* C16  = Vt + (size_t)BQ * HQ * DHQ * LQ;

  cvt_f32_to_f16<<<(unsigned)((N_HID + 255) / 256), 256, 0, stream>>>(hs, X16, (int)N_HID);
  cvt_f32_to_f16<<<(unsigned)((N_W + 255) / 256), 256, 0, stream>>>(Wq, Wq16, (int)N_W);
  cvt_f32_to_f16<<<(unsigned)((N_W + 255) / 256), 256, 0, stream>>>(Wk, Wk16, (int)N_W);
  cvt_f32_to_f16<<<(unsigned)((N_W + 255) / 256), 256, 0, stream>>>(Wv, Wv16, (int)N_W);
  cvt_f32_to_f16<<<(unsigned)((N_W + 255) / 256), 256, 0, stream>>>(Wo, Wo16, (int)N_W);

  resonance_fill<<<(BQ * HQ * LQ) / 256, 256, 0, stream>>>(phi, mag, mask, gamma, Qh, Kh);

  qkv_gemm<<<dim3(64, 8, 3), 256, 0, stream>>>(X16, Wq16, Wk16, Wv16,
                                               bq, bk, bv, Qh, Kh, Vt);

  flash_attn<<<dim3(16, 32), 256, 0, stream>>>(Qh, Kh, Vt, C16);

  out_proj_ln<<<dim3((BQ * LQ) / 16), 256, 0, stream>>>(C16, Wo16, bo, hs, lnw, lnb, out);
}
